// UniGCNII_29575144800476
// MI455X (gfx1250) — matrix operand order repaired and checked by reading, not compiled, not run on hardware
//
#include <hip/hip_runtime.h>
#include <math.h>

#define NNODES 50000
#define NEDGES 20000
#define NNZC   1000000
#define FIN    128
#define HID    64
#define NCLS   40
#define ALPHA  0.1f

typedef __attribute__((ext_vector_type(16))) _Float16 v16h;
typedef __attribute__((ext_vector_type(8)))  float    v8f;

__device__ __forceinline__ void atomic_add_f32(float* p, float v) {
  (void)__hip_atomic_fetch_add(p, v, __ATOMIC_RELAXED, __HIP_MEMORY_SCOPE_AGENT);
}

// ---------------- zero fill ----------------
__global__ void k_zero(float* __restrict__ p, int n) {
  int t = blockIdx.x * blockDim.x + threadIdx.x;
  if (t < n) p[t] = 0.0f;
}

// ---------------- degree counts (as float, recomputed each call) ----------------
__global__ void k_count(const int* __restrict__ vertex, const int* __restrict__ edges,
                        float* __restrict__ cnt_v, float* __restrict__ cnt_e) {
  int i = blockIdx.x * blockDim.x + threadIdx.x;
  if (i < NNZC) {
    atomic_add_f32(cnt_v + vertex[i], 1.0f);
    atomic_add_f32(cnt_e + edges[i], 1.0f);
  }
}

// ---------------- scatter-add: dst[didx[i]][:] += src[sidx[i]][:] (HID floats) ----
// 16 threads per nnz entry, each handles 4 consecutive floats (float4 load + 4 atomics)
__global__ void k_scatter(const float* __restrict__ src, const int* __restrict__ sidx,
                          const int* __restrict__ didx, float* __restrict__ dst) {
  int t = blockIdx.x * blockDim.x + threadIdx.x;
  int i = t >> 4;
  int c = (t & 15) * 4;
  if (i < NNZC) {
    int s = sidx[i], d = didx[i];
    const float4 v = *(const float4*)(src + s * HID + c);
    float* o = dst + d * HID + c;
    atomic_add_f32(o + 0, v.x);
    atomic_add_f32(o + 1, v.y);
    atomic_add_f32(o + 2, v.z);
    atomic_add_f32(o + 3, v.w);
  }
}

// ---------------- sum -> mean: mat[r][:] /= max(cnt[r],1) ----------------
__global__ void k_divcnt(float* __restrict__ mat, const float* __restrict__ cnt, int rows) {
  int t = blockIdx.x * blockDim.x + threadIdx.x;
  int r = t >> 4;
  int c = (t & 15) * 4;
  if (r < rows) {
    float inv = 1.0f / fmaxf(cnt[r], 1.0f);
    float4* p = (float4*)(mat + r * HID + c);
    float4 v = *p;
    v.x *= inv; v.y *= inv; v.z *= inv; v.w *= inv;
    *p = v;
  }
}

// ---------------- mean + L2 row-normalize + restart mix, one wave32 per node row ----
__global__ void k_finalize(float* __restrict__ Xi, const float* __restrict__ cnt_v,
                           const float* __restrict__ X0) {
  int gt   = blockIdx.x * blockDim.x + threadIdx.x;
  int row  = gt >> 5;            // one wave32 per row
  int lane = threadIdx.x & 31;   // 2 floats per lane (HID = 64)
  if (row >= NNODES) return;
  float inv = 1.0f / fmaxf(cnt_v[row], 1.0f);
  float2* rp = (float2*)(Xi + row * HID);
  float2 v = rp[lane];
  v.x *= inv; v.y *= inv;
  float ss = v.x * v.x + v.y * v.y;
#pragma unroll
  for (int m = 16; m >= 1; m >>= 1) ss += __shfl_xor(ss, m, 32);
  float nrm   = sqrtf(ss);
  float scale = (nrm > 0.0f) ? (1.0f / nrm) : 0.0f;
  const float2 x0 = ((const float2*)(X0 + row * HID))[lane];
  float2 o;
  o.x = (1.0f - ALPHA) * v.x * scale + ALPHA * x0.x;
  o.y = (1.0f - ALPHA) * v.y * scale + ALPHA * x0.y;
  rp[lane] = o;
}

// ---------------- generic WMMA GEMM, NT column tiles per wave ----------------
// out[m][n] = act( gscale * sum_k A[m][k]*W[k][n] + mixcoef*mix[m][n] + bias[n] )
// A: [M,K] row-major f32, W: [K,Nc] row-major f32, out/mix: [M,Nc] row-major.
// blockDim = (32,4): each wave owns one 16-row strip and ALL NT 16-col tiles:
// the A fragment is loaded/converted once per K-step and reused by NT WMMAs
// against B fragments read from the LDS-staged, f16, [n][k]-transposed W.
// K (template) must be a multiple of 32; M a multiple of 16 (50000 = 3125*16).
template <int K, int NT>
__global__ void k_wmma_gemm(const float* __restrict__ A, const float* __restrict__ W,
                            const float* __restrict__ bias, const float* __restrict__ mix,
                            float* __restrict__ out, float* __restrict__ out2,
                            int Nc, float gscale, float mixcoef, int relu, int mtiles) {
  __shared__ __align__(32) _Float16 wt[NT * 16 * K];   // W, [n][k] layout, zero-padded

  int tid = threadIdx.y * 32 + threadIdx.x;            // 0..127

  // ---- cooperative staging of all weight columns (zero-pad cols >= Nc) ----
#pragma unroll
  for (int idx = tid; idx < NT * 16 * K; idx += 128) {
    int n = idx / K;
    int k = idx - n * K;
    float w = (n < Nc) ? W[k * Nc + n] : 0.0f;
    wt[idx] = (_Float16)w;
  }
  __syncthreads();

  int rowTile = blockIdx.x * 4 + threadIdx.y;
  if (rowTile < mtiles) {                 // whole-wave uniform: EXEC all-ones for WMMA
    int row0 = rowTile * 16;
    int lane = threadIdx.x & 31;
    int m    = lane & 15;
    int h    = lane >> 4;

    const float* Arow = A + (row0 + m) * K;
    v8f acc[NT];
#pragma unroll
    for (int nt = 0; nt < NT; ++nt) acc[nt] = (v8f){};

#pragma unroll
    for (int kb = 0; kb < K; kb += 32) {
      // A fragment: lane needs K range [kb+16h, +16)
      const float4* ap0 = (const float4*)(Arow + kb + 16 * h);
      const float4* ap1 = (const float4*)(Arow + kb + 16 * h + 8);
      float4 a0 = ap0[0], a1 = ap0[1];
      float4 a2 = ap1[0], a3 = ap1[1];
      v16h a;
      a[0]  = (_Float16)a0.x;  a[1]  = (_Float16)a0.y;
      a[2]  = (_Float16)a0.z;  a[3]  = (_Float16)a0.w;
      a[4]  = (_Float16)a1.x;  a[5]  = (_Float16)a1.y;
      a[6]  = (_Float16)a1.z;  a[7]  = (_Float16)a1.w;
      a[8]  = (_Float16)a2.x;  a[9]  = (_Float16)a2.y;
      a[10] = (_Float16)a2.z;  a[11] = (_Float16)a2.w;
      a[12] = (_Float16)a3.x;  a[13] = (_Float16)a3.y;
      a[14] = (_Float16)a3.z;  a[15] = (_Float16)a3.w;
      // B fragment per column tile: 16 contiguous f16 (32B aligned) in LDS
#pragma unroll
      for (int nt = 0; nt < NT; ++nt) {
        v16h b = *(const v16h*)(wt + (nt * 16 + m) * K + kb + 16 * h);
        acc[nt] = __builtin_amdgcn_wmma_f32_16x16x32_f16(false, a, false, b, (short)0,
                                                         acc[nt], false, false);
      }
    }

    // C/D layout: lanes 0-15: N=lane, VGPR r -> M=r ; lanes 16-31: N=lane-16, M=r+8
#pragma unroll
    for (int nt = 0; nt < NT; ++nt) {
      int ncol = nt * 16 + m;
      if (ncol < Nc) {
#pragma unroll
        for (int r = 0; r < 8; ++r) {
          int mm = row0 + r + 8 * h;
          float v = acc[nt][r] * gscale;
          if (mix)  v += mixcoef * mix[mm * Nc + ncol];
          if (bias) v += bias[ncol];
          if (relu) v = fmaxf(v, 0.0f);
          out[mm * Nc + ncol] = v;
          if (out2) out2[mm * Nc + ncol] = v;
        }
      }
    }
  }
}

extern "C" void kernel_launch(void* const* d_in, const int* in_sizes, int n_in,
                              void* d_out, int out_size, void* d_ws, size_t ws_size,
                              hipStream_t stream) {
  (void)in_sizes; (void)n_in; (void)out_size; (void)ws_size;
  const float* x      = (const float*)d_in[0];   // [N, 128]
  const int*   vertex = (const int*)d_in[1];     // [NNZ]
  const int*   edges  = (const int*)d_in[2];     // [NNZ]
  const float* W0     = (const float*)d_in[3];   // [128, 64]
  const float* b0     = (const float*)d_in[4];   // [64]
  const float* Ws     = (const float*)d_in[5];   // [4, 64, 64]
  const float* Wout   = (const float*)d_in[6];   // [64, 40]
  const float* bout   = (const float*)d_in[7];   // [40]
  float*       out    = (float*)d_out;           // [N, 40]

  float* ws    = (float*)d_ws;
  float* X     = ws;                                   // [N, HID]
  float* X0    = X  + (size_t)NNODES * HID;            // [N, HID]
  float* Xi    = X0 + (size_t)NNODES * HID;            // [N, HID] (also Xv accumulator)
  float* Xe    = Xi + (size_t)NNODES * HID;            // [NE, HID]
  float* cnt_e = Xe + (size_t)NEDGES * HID;            // [NE]
  float* cnt_v = cnt_e + NEDGES;                       // [N]

  const int TPB = 256;
  // degree counts (cnt_e and cnt_v are contiguous -> one zero pass)
  int ncnt = NEDGES + NNODES;
  k_zero<<<(ncnt + TPB - 1) / TPB, TPB, 0, stream>>>(cnt_e, ncnt);
  k_count<<<(NNZC + TPB - 1) / TPB, TPB, 0, stream>>>(vertex, edges, cnt_v, cnt_e);

  const int mtiles = NNODES / 16;  // 3125
  dim3 gblk(32, 4);
  dim3 ggrid((mtiles + 3) / 4);

  // X = relu(x @ W0 + b0); X0 = X
  k_wmma_gemm<FIN, 4><<<ggrid, gblk, 0, stream>>>(x, W0, b0, nullptr, X, X0,
                                                  HID, 1.0f, 0.0f, 1, mtiles);

  const int nScatter = (NNZC * 16 + TPB - 1) / TPB;
  for (int l = 0; l < 4; ++l) {
    float beta = logf(0.5f / (float)(l + 1) + 1.0f);
    // Xe = scatter_mean(X[vertex], edges)
    k_zero<<<(NEDGES * HID + TPB - 1) / TPB, TPB, 0, stream>>>(Xe, NEDGES * HID);
    k_scatter<<<nScatter, TPB, 0, stream>>>(X, vertex, edges, Xe);
    k_divcnt<<<(NEDGES * 16 + TPB - 1) / TPB, TPB, 0, stream>>>(Xe, cnt_e, NEDGES);
    // Xv = scatter_mean(Xe[edges], vertex); normalize; Xi = 0.9*Xv + 0.1*X0
    k_zero<<<(NNODES * HID + TPB - 1) / TPB, TPB, 0, stream>>>(Xi, NNODES * HID);
    k_scatter<<<nScatter, TPB, 0, stream>>>(Xe, edges, vertex, Xi);
    k_finalize<<<(NNODES * 32 + TPB - 1) / TPB, TPB, 0, stream>>>(Xi, cnt_v, X0);
    // X = relu((1-beta)*Xi + beta*(Xi @ Ws[l]))
    k_wmma_gemm<HID, 4><<<ggrid, gblk, 0, stream>>>(Xi, Ws + (size_t)l * HID * HID,
                                                    nullptr, Xi, X, nullptr, HID, beta,
                                                    1.0f - beta, 1, mtiles);
  }

  // out = X @ Wout + bout (40 cols -> 3 column tiles, store-guarded)
  k_wmma_gemm<HID, 3><<<ggrid, gblk, 0, stream>>>(X, Wout, bout, nullptr, out, nullptr,
                                                  NCLS, 1.0f, 0.0f, 0, mtiles);
}
